// BinaryNormalizedLinear_45200235823119
// MI455X (gfx1250) — matrix operand order repaired and checked by reading, not compiled, not run on hardware
//
#include <hip/hip_runtime.h>
#include <math.h>

typedef __attribute__((ext_vector_type(16))) _Float16 v16h;
typedef __attribute__((ext_vector_type(8)))  _Float16 v8h;
typedef __attribute__((ext_vector_type(8)))  float    v8f;

#define EPS 1e-8f

// ------------------------------------------------------------------
// deterministic block reduction (fixed tree, no atomics)
// ------------------------------------------------------------------
__device__ __forceinline__ float block_reduce_sum_256(float v, float* smem) {
    int tid = threadIdx.x;
    smem[tid] = v;
    __syncthreads();
    #pragma unroll
    for (int s = 128; s > 0; s >>= 1) {
        if (tid < s) smem[tid] += smem[tid + s];
        __syncthreads();
    }
    return smem[0];
}

__global__ __launch_bounds__(256)
void k_partial_sum(const float* __restrict__ in, long long n, float* __restrict__ partials) {
    __shared__ float smem[256];
    float acc = 0.f;
    long long stride = (long long)gridDim.x * 256;
    for (long long i = (long long)blockIdx.x * 256 + threadIdx.x; i < n; i += stride)
        acc += in[i];
    float s = block_reduce_sum_256(acc, smem);
    if (threadIdx.x == 0) partials[blockIdx.x] = s;
}

__global__ __launch_bounds__(256)
void k_finalize_mean(const float* __restrict__ partials, int np, float inv_n,
                     float* __restrict__ out) {
    __shared__ float smem[256];
    float acc = 0.f;
    for (int i = threadIdx.x; i < np; i += 256) acc += partials[i];
    float s = block_reduce_sum_256(acc, smem);
    if (threadIdx.x == 0) out[0] = s * inv_n;
}

// ------------------------------------------------------------------
// conversions / straight-through binary quantization
// ------------------------------------------------------------------
__global__ __launch_bounds__(256)
void k_cvt_x_f16(const float* __restrict__ x, _Float16* __restrict__ xh, long long n) {
    long long stride = (long long)gridDim.x * 256;
    for (long long i = (long long)blockIdx.x * 256 + threadIdx.x; i < n; i += stride)
        xh[i] = (_Float16)x[i];
}

__global__ __launch_bounds__(256)
void k_quant_w(const float* __restrict__ W, const float* __restrict__ meanW,
               _Float16* __restrict__ wh, long long n) {
    float m = *meanW;
    long long stride = (long long)gridDim.x * 256;
    for (long long i = (long long)blockIdx.x * 256 + threadIdx.x; i < n; i += stride)
        wh[i] = (W[i] > m) ? (_Float16)1.0f : (_Float16)0.0f;
}

__global__ __launch_bounds__(256)
void k_quant_b(const float* __restrict__ b, const float* __restrict__ meanB,
               float* __restrict__ bq, int n) {
    float m = *meanB;
    int i = blockIdx.x * 256 + threadIdx.x;
    if (i < n) bq[i] = (b[i] > m) ? 1.0f : 0.0f;
}

// ------------------------------------------------------------------
// fragment loaders / MMA helpers (compile-time indices only)
// ------------------------------------------------------------------
__device__ __forceinline__ void load_afrags(v16h af[4], const _Float16* const aptr[4], int kb) {
    // 16-bit A 16x32: lane lr holds row M=lr;
    // halves 0..7 = K = ksel*8 + 0..7, halves 8..15 = K = ksel*8 + 16..23
    #pragma unroll
    for (int i = 0; i < 4; ++i) {
        v8h lo = *(const v8h*)(aptr[i] + kb);
        v8h hi = *(const v8h*)(aptr[i] + kb + 16);
        #pragma unroll
        for (int t = 0; t < 8; ++t) { af[i][t] = lo[t]; af[i][t + 8] = hi[t]; }
    }
}

__device__ __forceinline__ void load_bfrags(v16h bf[4], const _Float16* const bptr[4], int kb) {
    // 16-bit B 32x16: lane lr holds column N=lr;
    // halves 0..7 = K = ksel*8 + 0..7, halves 8..15 = K = ksel*8 + 16..23
    #pragma unroll
    for (int j = 0; j < 4; ++j) {
        v8h lo = *(const v8h*)(bptr[j] + kb);
        v8h hi = *(const v8h*)(bptr[j] + kb + 16);
        #pragma unroll
        for (int t = 0; t < 8; ++t) { bf[j][t] = lo[t]; bf[j][t + 8] = hi[t]; }
    }
}

__device__ __forceinline__ void mma16(v8f acc[4][4], const v16h af[4], const v16h bf[4]) {
    #pragma unroll
    for (int i = 0; i < 4; ++i)
        #pragma unroll
        for (int j = 0; j < 4; ++j)
            acc[i][j] = __builtin_amdgcn_wmma_f32_16x16x32_f16(
                /*neg_a=*/false, af[i], /*neg_b=*/false, bf[j],
                /*c_mod=*/(short)0, acc[i][j],
                /*reuse_a=*/false, /*reuse_b=*/false);
}

// ------------------------------------------------------------------
// WMMA GEMM: z[M,N] = xh[M,K] * wh[N,K]^T  (f16 inputs, f32 accum)
// 256 threads = 8 waves (2 along M x 4 along N), block tile 128x256.
// Wave tile 64x64 = 4x4 fragments; K unrolled by 2 (two K=32 slices)
// with explicit double-buffered A/B fragments so the 16 loads of one
// slice are in flight while the 16 v_wmma of the other slice execute.
// Requires K % 64 == 0.
// ------------------------------------------------------------------
__global__ __launch_bounds__(256)
void k_gemm_wmma(const _Float16* __restrict__ xh, const _Float16* __restrict__ wh,
                 float* __restrict__ z, int M, int N, int K) {
    const int lane = threadIdx.x & 31;
    const int wave = threadIdx.x >> 5;
    const int wm   = wave & 1;        // 0..1 : 64-row strip within block
    const int wn   = wave >> 1;       // 0..3 : 64-col strip within block
    const int bm   = blockIdx.x * 128;
    const int bn   = blockIdx.y * 256;
    const int lr   = lane & 15;
    const int ksel = lane >> 4;

    v8f acc[4][4] = {};

    const _Float16* aptr[4];
    #pragma unroll
    for (int i = 0; i < 4; ++i)
        aptr[i] = xh + (size_t)(bm + wm * 64 + i * 16 + lr) * K + ksel * 8;

    const _Float16* bptr[4];
    #pragma unroll
    for (int j = 0; j < 4; ++j)
        bptr[j] = wh + (size_t)(bn + wn * 64 + j * 16 + lr) * K + ksel * 8;

    v16h a0[4], b0[4], a1[4], b1[4];
    load_afrags(a0, aptr, 0);
    load_bfrags(b0, bptr, 0);

    for (int kb = 0; kb < K; kb += 64) {
        // slice 1 loads in flight while slice 0 computes
        load_afrags(a1, aptr, kb + 32);
        load_bfrags(b1, bptr, kb + 32);
        __builtin_prefetch(aptr[0] + kb + 64, 0, 1);   // global_prefetch_b8
        __builtin_prefetch(bptr[0] + kb + 64, 0, 1);
        mma16(acc, a0, b0);

        if (kb + 64 < K) {
            load_afrags(a0, aptr, kb + 64);
            load_bfrags(b0, bptr, kb + 64);
        }
        mma16(acc, a1, b1);
    }

    // C layout: 8 VGPRs; lanes 0-15: N=lr, M=r; lanes 16-31: N=lr, M=r+8
    #pragma unroll
    for (int i = 0; i < 4; ++i)
        #pragma unroll
        for (int j = 0; j < 4; ++j) {
            int row0 = bm + wm * 64 + i * 16 + ksel * 8;
            int col  = bn + wn * 64 + j * 16 + lr;
            #pragma unroll
            for (int r = 0; r < 8; ++r)
                z[(size_t)(row0 + r) * N + col] = acc[i][j][r];
        }
}

// ------------------------------------------------------------------
// row normalization: z = (z + bq - mean) / (std + eps), ddof=1, in-place
// one block per row, N = 4096 -> 16 elems/thread kept in registers
// ------------------------------------------------------------------
__global__ __launch_bounds__(256)
void k_rownorm(float* __restrict__ z, const float* __restrict__ bq,
               int N, float inv_n, float inv_nm1) {
    __shared__ float smem[256];
    float* zr = z + (size_t)blockIdx.x * N;
    const int T = N / 256;   // 16 for N=4096
    float v[16];
    float s = 0.f, ss = 0.f;
    for (int t = 0; t < T; ++t) {
        int idx = t * 256 + threadIdx.x;
        float val = zr[idx] + bq[idx];
        v[t] = val;
        s  += val;
        ss += val * val;
    }
    float tot = block_reduce_sum_256(s, smem);
    __syncthreads();
    float tot2 = block_reduce_sum_256(ss, smem);
    float mean = tot * inv_n;
    float var  = (tot2 - tot * tot * inv_n) * inv_nm1;
    var = var < 0.f ? 0.f : var;
    float inv = 1.f / (sqrtf(var) + EPS);
    for (int t = 0; t < T; ++t) {
        int idx = t * 256 + threadIdx.x;
        zr[idx] = (v[t] - mean) * inv;
    }
}

// ------------------------------------------------------------------
extern "C" void kernel_launch(void* const* d_in, const int* in_sizes, int n_in,
                              void* d_out, int out_size, void* d_ws, size_t ws_size,
                              hipStream_t stream) {
    const float* x = (const float*)d_in[0];
    const float* W = (const float*)d_in[1];
    const float* b = (const float*)d_in[2];
    float* z = (float*)d_out;

    const int OUT = in_sizes[2];             // 4096
    const int IN  = in_sizes[1] / OUT;       // 4096
    const int M   = in_sizes[0] / IN;        // 8192
    const long long nW = (long long)in_sizes[1];
    const long long nX = (long long)in_sizes[0];

    // workspace layout
    char* ws = (char*)d_ws;
    float* meanW = (float*)(ws + 0);
    float* meanB = (float*)(ws + 16);
    float* partW = (float*)(ws + 256);                  // 1024 floats
    float* partB = (float*)(ws + 256 + 4096);           // 64 floats
    float* bq    = (float*)(ws + 16384);                // OUT floats
    size_t off = 16384 + (size_t)OUT * sizeof(float);
    off = (off + 255) & ~(size_t)255;
    _Float16* xh = (_Float16*)(ws + off);
    off += (size_t)M * IN * sizeof(_Float16);
    off = (off + 255) & ~(size_t)255;
    _Float16* wh = (_Float16*)(ws + off);

    // 1) global means of W and b (deterministic two-pass tree)
    k_partial_sum<<<1024, 256, 0, stream>>>(W, nW, partW);
    k_finalize_mean<<<1, 256, 0, stream>>>(partW, 1024, 1.f / (float)nW, meanW);
    k_partial_sum<<<64, 256, 0, stream>>>(b, (long long)OUT, partB);
    k_finalize_mean<<<1, 256, 0, stream>>>(partB, 64, 1.f / (float)OUT, meanB);

    // 2) convert / quantize
    k_cvt_x_f16<<<4096, 256, 0, stream>>>(x, xh, nX);
    k_quant_w<<<2048, 256, 0, stream>>>(W, meanW, wh, nW);
    k_quant_b<<<(OUT + 255) / 256, 256, 0, stream>>>(b, meanB, bq, OUT);

    // 3) WMMA GEMM z = xh * wh^T  (block tile 128x256, K%64==0)
    dim3 grid(M / 128, OUT / 256);
    k_gemm_wmma<<<grid, 256, 0, stream>>>(xh, wh, z, M, OUT, IN);

    // 4) fused +bq and per-row standardization (in-place on d_out)
    k_rownorm<<<M, 256, 0, stream>>>(z, bq, OUT, 1.f / (float)OUT,
                                     1.f / (float)(OUT - 1));
}
